// TactileNetwork_27401891348690
// MI455X (gfx1250) — matrix operand order repaired and checked by reading, not compiled, not run on hardware
//
#include <hip/hip_runtime.h>

// ---------------------------------------------------------------------------
// Types for CDNA5 WMMA (wave32, 16x16x32 bf16 -> f32)
// ---------------------------------------------------------------------------
typedef __bf16 bf16_t;
typedef bf16_t v8bf  __attribute__((ext_vector_type(8)));
typedef bf16_t v16bf __attribute__((ext_vector_type(16)));
typedef float  v8f   __attribute__((ext_vector_type(8)));

// ---------------------------------------------------------------------------
// Pack conv weights [OC][IC][4][4] fp32 -> [OC][tap][IC] bf16  (tap = ky*4+kx)
// Each 32-wide K-step of the implicit GEMM becomes a contiguous channel run.
// ---------------------------------------------------------------------------
__global__ __launch_bounds__(256)
void pack_w_kernel(const float* __restrict__ w, bf16_t* __restrict__ wp,
                   int OC, int IC) {
    int i = blockIdx.x * 256 + threadIdx.x;
    int total = OC * 16 * IC;
    if (i >= total) return;
    int ic  = i % IC;
    int rem = i / IC;
    int tap = rem % 16;
    int oc  = rem / 16;
    int ky = tap >> 2, kx = tap & 3;
    wp[i] = (bf16_t)w[(((size_t)oc * IC + ic) * 4 + ky) * 4 + kx];
}

// ---------------------------------------------------------------------------
// Zero the 1-pixel halo ring of a padded NHWC bf16 activation buffer
// buf: [N][HP][WP][C], zero rows 0 / HP-1 and cols 0 / WP-1. One block per n.
// ---------------------------------------------------------------------------
__global__ __launch_bounds__(256)
void halo_zero_kernel(bf16_t* __restrict__ buf, int HP, int WP, int C) {
    int n = blockIdx.x;
    int nbp = 2 * WP + 2 * (HP - 2);
    v8bf z = {};
    for (int i = threadIdx.x; i < nbp; i += 256) {
        int r, c;
        if (i < WP)            { r = 0;                      c = i; }
        else if (i < 2 * WP)   { r = HP - 1;                 c = i - WP; }
        else if (i < 2 * WP + (HP - 2)) { r = i - 2 * WP + 1;          c = 0; }
        else                   { r = i - 2 * WP - (HP - 2) + 1; c = WP - 1; }
        bf16_t* p = buf + (((size_t)n * HP + r) * WP + c) * C;
        for (int j = 0; j < C; j += 8) *(v8bf*)(p + j) = z;
    }
}

// ---------------------------------------------------------------------------
// conv1: 1->32 ch, 4x4 stride-2 pad-1, 256x256 -> 128x128, ReLU.
// Bandwidth bound (K=16). Direct VALU conv, weights in LDS, output written as
// packed bf16 NHWC into the INTERIOR of the padded h1 buffer [128][130][130][32].
// ---------------------------------------------------------------------------
__global__ __launch_bounds__(256)
void conv1_kernel(const float* __restrict__ x, const float* __restrict__ w1,
                  const float* __restrict__ b1, bf16_t* __restrict__ h1p) {
    __shared__ float ws[512];
    __shared__ float bs[32];
    for (int i = threadIdx.x; i < 512; i += 256) ws[i] = w1[i];
    if (threadIdx.x < 32) bs[threadIdx.x] = b1[threadIdx.x];
    __syncthreads();

    long pix = (long)blockIdx.x * 256 + threadIdx.x;   // 128*128*128 total
    int n   = (int)(pix >> 14);
    int rem = (int)(pix & 16383);
    int y = rem >> 7, xo = rem & 127;

    float xv[16];
#pragma unroll
    for (int ky = 0; ky < 4; ++ky)
#pragma unroll
        for (int kx = 0; kx < 4; ++kx) {
            int iy = 2 * y - 1 + ky, ix = 2 * xo - 1 + kx;
            float v = 0.0f;
            if (iy >= 0 && iy < 256 && ix >= 0 && ix < 256)
                v = x[((size_t)n * 256 + iy) * 256 + ix];
            xv[ky * 4 + kx] = v;
        }

    bf16_t* dst = h1p + (((size_t)n * 130 + (y + 1)) * 130 + (xo + 1)) * 32;
#pragma unroll
    for (int g = 0; g < 4; ++g) {
        v8bf o;
#pragma unroll
        for (int r = 0; r < 8; ++r) {
            int oc = g * 8 + r;
            float acc = bs[oc];
#pragma unroll
            for (int t = 0; t < 16; ++t) acc += ws[oc * 16 + t] * xv[t];
            acc = acc > 0.0f ? acc : 0.0f;
            o[r] = (bf16_t)acc;
        }
        *(v8bf*)(dst + g * 8) = o;
    }
}

// ---------------------------------------------------------------------------
// WMMA implicit-GEMM stride-2 4x4 conv + bias + ReLU over PADDED bf16 NHWC
// input [N][HIN+2][WIN+2][CIN] -> output [N][HOP][WOP][COUT] (PADO halo).
//   M = COUT, N = B*HOUT*WOUT pixels, K = 16*CIN.
// Halo padding in memory => branch-free inner loop with all tap offsets folded
// into load immediates. Fully unrolled, DISTANCE-2 software pipeline with a
// 3-slot rotating buffer: step s consumes slot s%3 while slot (s+2)%3 is being
// filled, so >= one full K-step of loads is always in flight when a WMMA
// issues and the buffers cannot be register-coalesced away. Each wave:
// 16 oc x 64 px (A fragment reused across the 4 pixel subtiles).
// LDSW: stage the whole packed weight matrix in LDS (conv2: 64 KB).
// ---------------------------------------------------------------------------
template <int CIN, int COUT, int HOUT, int WOUT, int WAVES_OC, int WAVES_PX,
          bool LDSW, int HOP, int WOP, int PADO>
__global__ __launch_bounds__(256)
void conv_s2_wmma(const bf16_t* __restrict__ hin, const bf16_t* __restrict__ wp,
                  const float* __restrict__ bias, bf16_t* __restrict__ hout) {
    static_assert(WAVES_OC * WAVES_PX == 8, "8 waves per block");
    constexpr int WINP = 2 * WOUT + 2;                 // padded input width
    constexpr int HINP = 2 * HOUT + 2;
    constexpr int K = 16 * CIN;
    constexpr int NKSTEP = K / 32;

    extern __shared__ char smem_raw[];
    bf16_t* wlds = (bf16_t*)smem_raw;
    if (LDSW) {
        const uint4* src = (const uint4*)wp;
        uint4* dst = (uint4*)smem_raw;
        for (int i = threadIdx.x; i < COUT * K / 8; i += 256) dst[i] = src[i];
        __syncthreads();
    }

    const int tid   = threadIdx.x;
    const int lane  = tid & 31;
    const int wave  = tid >> 5;
    const int col   = lane & 15;
    const int khalf = lane >> 4;

    const int oc_tile = wave % WAVES_OC;
    const int px_grp  = wave / WAVES_OC;
    const int ocbase  = oc_tile * 16;
    const long wave_px0 = (long)blockIdx.x * (64 * WAVES_PX) + (long)px_grp * 64;

    // Per-subtile coords (16|WOUT, each 16-px subtile sits in one output row).
    int n_t[4], y_t[4], x0_t[4];
    const bf16_t* bbase[4];                            // loop-invariant lane ptrs
#pragma unroll
    for (int t = 0; t < 4; ++t) {
        long p = wave_px0 + t * 16;
        n_t[t]  = (int)(p / (HOUT * WOUT));
        int rem = (int)(p % (HOUT * WOUT));
        y_t[t]  = rem / WOUT;
        x0_t[t] = rem % WOUT;
        // padded coords: iy = 2y + ky, ix = 2(x0+col) + kx  (always in range)
        bbase[t] = hin + (((size_t)n_t[t] * HINP + 2 * y_t[t]) * WINP +
                          2 * (x0_t[t] + col)) * CIN + khalf * 16;
    }

    const bf16_t* arow = (LDSW ? (const bf16_t*)wlds : wp) + (size_t)(ocbase + col) * K;
    const int aoff = khalf * 16;

    // A fragment in contiguous k order (k = 16*khalf + s, same as B): two 16B chunks + concat.
    auto load_a = [&](int s) -> v16bf {
        const bf16_t* ap = arow + s * 32;
        v8bf alo = *(const v8bf*)(ap + aoff);
        v8bf ahi = *(const v8bf*)(ap + aoff + 8);
        return __builtin_shufflevector(alo, ahi, 0, 1, 2, 3, 4, 5, 6, 7,
                                       8, 9, 10, 11, 12, 13, 14, 15);
    };
    // Tap / channel-block offset for K-step s (compile-time under full unroll).
    auto b_off = [&](int s) -> int {
        const int tap = (s * 32) / CIN;
        const int ky = tap >> 2, kx = tap & 3;
        const int icoff = (CIN == 64) ? (s & 1) * 32 : 0;
        return (ky * WINP + kx) * CIN + icoff;
    };

    v8f acc[4] = {{}, {}, {}, {}};

    // ---- distance-2 software pipeline, 3-slot rotating buffers ----
    v16bf A[3];
    v16bf B[3][4];
#pragma unroll
    for (int s = 0; s < 2 && s < NKSTEP; ++s) {        // prologue: stages 0,1
        A[s] = load_a(s);
        const int bo = b_off(s);
#pragma unroll
        for (int t = 0; t < 4; ++t) B[s][t] = *(const v16bf*)(bbase[t] + bo);
    }

#pragma unroll
    for (int s = 0; s < NKSTEP; ++s) {
        const int cs = s % 3;                          // slot consumed now
        const int fs = (s + 2) % 3;                    // slot being filled
        if (s + 2 < NKSTEP) {                          // compile-time branch
            A[fs] = load_a(s + 2);
            const int bo = b_off(s + 2);
#pragma unroll
            for (int t = 0; t < 4; ++t) B[fs][t] = *(const v16bf*)(bbase[t] + bo);
        }
#pragma unroll
        for (int t = 0; t < 4; ++t)
            acc[t] = __builtin_amdgcn_wmma_f32_16x16x32_bf16(
                false, A[cs], false, B[cs][t], (short)0, acc[t], false, false);
    }

    // ---- epilogue: bias + ReLU + bf16 pack, one v8bf store per subtile ----
    const int oc0 = ocbase + khalf * 8;                // D: lane>=16 holds M=8..15
    v8f bv = *(const v8f*)(bias + oc0);
#pragma unroll
    for (int t = 0; t < 4; ++t) {
        v8bf o;
#pragma unroll
        for (int r = 0; r < 8; ++r) {
            float v = acc[t][r] + bv[r];
            v = v > 0.0f ? v : 0.0f;
            o[r] = (bf16_t)v;
        }
        bf16_t* dst = hout + (((size_t)n_t[t] * HOP + y_t[t] + PADO) * WOP +
                              x0_t[t] + col + PADO) * COUT + oc0;
        *(v8bf*)dst = o;
    }
}

// ---------------------------------------------------------------------------
// Global mean pool: h3 [n][32*32][128] bf16 -> h [n][128] fp32
// ---------------------------------------------------------------------------
__global__ __launch_bounds__(256)
void meanpool_kernel(const bf16_t* __restrict__ h3, float* __restrict__ h) {
    __shared__ float s[256];
    int n = blockIdx.x;
    int c = threadIdx.x & 127;
    int half = threadIdx.x >> 7;
    const bf16_t* base = h3 + ((size_t)n * 1024 + (size_t)half * 512) * 128 + c;
    float acc = 0.0f;
    for (int p = 0; p < 512; ++p) acc += (float)base[(size_t)p * 128];
    s[threadIdx.x] = acc;
    __syncthreads();
    if (threadIdx.x < 128)
        h[n * 128 + threadIdx.x] = (s[threadIdx.x] + s[threadIdx.x + 128]) * (1.0f / 1024.0f);
}

// z_e[n][e] = b_lat[e] + h[n][:] . W_lat[e][:]
__global__ __launch_bounds__(256)
void latent_kernel(const float* __restrict__ h, const float* __restrict__ Wlat,
                   const float* __restrict__ blat, float* __restrict__ ze) {
    __shared__ float hs[128];
    int n = blockIdx.x;
    if (threadIdx.x < 128) hs[threadIdx.x] = h[n * 128 + threadIdx.x];
    __syncthreads();
    int e = threadIdx.x;
    float acc = blat[e];
    const float* wr = Wlat + (size_t)e * 128;
#pragma unroll 4
    for (int c = 0; c < 128; ++c) acc += hs[c] * wr[c];
    ze[n * 256 + e] = acc;
}

// ---------------------------------------------------------------------------
// VQ: nearest codebook row (argmin over 128 codes, dim 256); writes z_q and a
// per-row squared-error partial for the deterministic loss reduction.
// ---------------------------------------------------------------------------
__global__ __launch_bounds__(128)
void vq_kernel(const float* __restrict__ ze, const float* __restrict__ cb,
               float* __restrict__ zq, float* __restrict__ lpart) {
    __shared__ float zs[256];
    __shared__ float dist[128];
    __shared__ int   idxs[128];
    int n = blockIdx.x, t = threadIdx.x;
    zs[t] = ze[n * 256 + t];
    zs[t + 128] = ze[n * 256 + 128 + t];
    __syncthreads();

    const float* cr = cb + (size_t)t * 256;
    float d = 0.0f;
    for (int e = 0; e < 256; ++e) {
        float diff = zs[e] - cr[e];
        d += diff * diff;
    }
    dist[t] = d;
    idxs[t] = t;
    __syncthreads();
    for (int sft = 64; sft > 0; sft >>= 1) {
        if (t < sft) {
            float da = dist[t], db = dist[t + sft];
            int ia = idxs[t], ib = idxs[t + sft];
            if (db < da || (db == da && ib < ia)) { dist[t] = db; idxs[t] = ib; }
        }
        __syncthreads();
    }
    int best = idxs[0];
    __syncthreads();

    const float* cbest = cb + (size_t)best * 256;
    float q0 = cbest[t], q1 = cbest[t + 128];
    zq[n * 256 + t] = q0;
    zq[n * 256 + 128 + t] = q1;
    float d0 = q0 - zs[t], d1 = q1 - zs[t + 128];
    dist[t] = d0 * d0 + d1 * d1;
    __syncthreads();
    for (int sft = 64; sft > 0; sft >>= 1) {
        if (t < sft) dist[t] += dist[t + sft];
        __syncthreads();
    }
    if (t == 0) lpart[n] = dist[0];
}

__global__ __launch_bounds__(128)
void loss_reduce_kernel(const float* __restrict__ lpart, float* __restrict__ loss) {
    __shared__ float s[128];
    int t = threadIdx.x;
    s[t] = lpart[t];
    __syncthreads();
    for (int sft = 64; sft > 0; sft >>= 1) {
        if (t < sft) s[t] += s[t + sft];
        __syncthreads();
    }
    if (t == 0) *loss = s[0] * (1.0f / 32768.0f);
}

// z_d[n][j] = b_proj[j] + z_q[n][:] . W_proj[j][:]
__global__ __launch_bounds__(128)
void proj_kernel(const float* __restrict__ zq, const float* __restrict__ Wproj,
                 const float* __restrict__ bproj, float* __restrict__ zd) {
    __shared__ float qs[256];
    int n = blockIdx.x, t = threadIdx.x;
    qs[t] = zq[n * 256 + t];
    qs[t + 128] = zq[n * 256 + 128 + t];
    __syncthreads();
    float acc = bproj[t];
    const float* wr = Wproj + (size_t)t * 256;
#pragma unroll 4
    for (int e = 0; e < 256; ++e) acc += qs[e] * wr[e];
    zd[n * 128 + t] = acc;
}

// ---------------------------------------------------------------------------
// Transposed convs (stride 2, k4, pad 1): tap relation k = o - 2*i + 1.
// Stage 1: 1x1 -> 2x2 (only i=0 contributes, k = o+1).
// ---------------------------------------------------------------------------
__global__ __launch_bounds__(256)
void dec1_kernel(const float* __restrict__ zd, const float* __restrict__ d1,
                 const float* __restrict__ bd1, float* __restrict__ g1) {
    __shared__ float zs[128];
    int n = blockIdx.x, t = threadIdx.x;
    if (t < 128) zs[t] = zd[n * 128 + t];
    __syncthreads();
    int oc = t >> 2, oy = (t >> 1) & 1, ox = t & 1;
    const float* wbase = d1 + (size_t)oc * 16 + (oy + 1) * 4 + (ox + 1);
    float acc = bd1[oc];
    for (int ic = 0; ic < 128; ++ic) acc += zs[ic] * wbase[(size_t)ic * 1024];
    acc = acc > 0.0f ? acc : 0.0f;
    g1[((size_t)n * 64 + oc) * 4 + oy * 2 + ox] = acc;
}

// Stage 2: 2x2 -> 4x4, 64 -> 32 ch
__global__ __launch_bounds__(256)
void dec2_kernel(const float* __restrict__ g1, const float* __restrict__ d2,
                 const float* __restrict__ bd2, float* __restrict__ g2) {
    __shared__ float gs[256];   // [ic(64)][iy(2)][ix(2)]
    int n = blockIdx.x, t = threadIdx.x;
    gs[t] = g1[(size_t)n * 256 + t];
    __syncthreads();
    for (int o = t; o < 512; o += 256) {
        int oc = o >> 4, oy = (o >> 2) & 3, ox = o & 3;
        float acc = bd2[oc];
        for (int ic = 0; ic < 64; ++ic) {
            const float* wr = d2 + ((size_t)ic * 32 + oc) * 16;
#pragma unroll
            for (int iy = 0; iy < 2; ++iy) {
                int ky = oy - 2 * iy + 1;
                if (ky < 0 || ky > 3) continue;
#pragma unroll
                for (int ix = 0; ix < 2; ++ix) {
                    int kx = ox - 2 * ix + 1;
                    if (kx < 0 || kx > 3) continue;
                    acc += gs[ic * 4 + iy * 2 + ix] * wr[ky * 4 + kx];
                }
            }
        }
        acc = acc > 0.0f ? acc : 0.0f;
        g2[((size_t)n * 32 + oc) * 16 + oy * 4 + ox] = acc;
    }
}

// Stage 3: 4x4 -> 8x8, 32 -> 1 ch (no ReLU)
__global__ __launch_bounds__(64)
void dec3_kernel(const float* __restrict__ g2, const float* __restrict__ d3,
                 const float* __restrict__ bd3, float* __restrict__ xr) {
    __shared__ float gs[512];
    int n = blockIdx.x, t = threadIdx.x;
    for (int i = t; i < 512; i += 64) gs[i] = g2[(size_t)n * 512 + i];
    __syncthreads();
    int oy = t >> 3, ox = t & 7;
    float acc = bd3[0];
    for (int ic = 0; ic < 32; ++ic) {
        const float* wr = d3 + (size_t)ic * 16;
#pragma unroll
        for (int iy = 0; iy < 4; ++iy) {
            int ky = oy - 2 * iy + 1;
            if (ky < 0 || ky > 3) continue;
#pragma unroll
            for (int ix = 0; ix < 4; ++ix) {
                int kx = ox - 2 * ix + 1;
                if (kx < 0 || kx > 3) continue;
                acc += gs[ic * 16 + iy * 4 + ix] * wr[ky * 4 + kx];
            }
        }
    }
    xr[(size_t)n * 64 + t] = acc;
}

// ---------------------------------------------------------------------------
// Launcher
// ---------------------------------------------------------------------------
extern "C" void kernel_launch(void* const* d_in, const int* in_sizes, int n_in,
                              void* d_out, int out_size, void* d_ws, size_t ws_size,
                              hipStream_t stream) {
    const float* x     = (const float*)d_in[0];
    const float* w1    = (const float*)d_in[1];
    const float* b1    = (const float*)d_in[2];
    const float* w2    = (const float*)d_in[3];
    const float* b2    = (const float*)d_in[4];
    const float* w3    = (const float*)d_in[5];
    const float* b3    = (const float*)d_in[6];
    const float* Wlat  = (const float*)d_in[7];
    const float* blat  = (const float*)d_in[8];
    const float* cb    = (const float*)d_in[9];
    const float* Wproj = (const float*)d_in[10];
    const float* bproj = (const float*)d_in[11];
    const float* d1    = (const float*)d_in[12];
    const float* bd1   = (const float*)d_in[13];
    const float* d2    = (const float*)d_in[14];
    const float* bd2   = (const float*)d_in[15];
    const float* d3    = (const float*)d_in[16];
    const float* bd3   = (const float*)d_in[17];

    float* out  = (float*)d_out;
    float* xr   = out;           // 128*1*8*8   = 8192
    float* ze   = out + 8192;    // 128*256     = 32768
    float* zq   = out + 40960;   // 128*256     = 32768
    float* loss = out + 73728;   // scalar

    char* ws = (char*)d_ws;
    size_t off = 0;
    auto alloc = [&](size_t bytes) -> char* {
        char* p = ws + off;
        off += (bytes + 255) & ~(size_t)255;
        return p;
    };
    bf16_t* h1p   = (bf16_t*)alloc((size_t)128 * 130 * 130 * 32 * 2); // 138 MB padded
    bf16_t* h2p   = (bf16_t*)alloc((size_t)128 * 66 * 66 * 64 * 2);   //  71 MB padded
    bf16_t* h3    = (bf16_t*)alloc((size_t)128 * 32 * 32 * 128 * 2);  //  34 MB
    bf16_t* wp2   = (bf16_t*)alloc((size_t)64 * 512 * 2);
    bf16_t* wp3   = (bf16_t*)alloc((size_t)128 * 1024 * 2);
    float*  h     = (float*)alloc((size_t)128 * 128 * 4);
    float*  zd    = (float*)alloc((size_t)128 * 128 * 4);
    float*  g1    = (float*)alloc((size_t)128 * 64 * 4 * 4);
    float*  g2    = (float*)alloc((size_t)128 * 32 * 16 * 4);
    float*  lpart = (float*)alloc((size_t)128 * 4);

    // Weight packing + halo zeroing (tiny, overlap-friendly)
    pack_w_kernel<<<(64 * 16 * 32 + 255) / 256, 256, 0, stream>>>(w2, wp2, 64, 32);
    pack_w_kernel<<<(128 * 16 * 64 + 255) / 256, 256, 0, stream>>>(w3, wp3, 128, 64);
    halo_zero_kernel<<<128, 256, 0, stream>>>(h1p, 130, 130, 32);
    halo_zero_kernel<<<128, 256, 0, stream>>>(h2p, 66, 66, 64);

    // Encoder
    conv1_kernel<<<8192, 256, 0, stream>>>(x, w1, b1, h1p);
    // conv2: 64 oc, 524288 px, 128 px/block -> 4096 blocks, 64 KB LDS weights,
    // writes interior of padded h2
    conv_s2_wmma<32, 64, 64, 64, 4, 2, true, 66, 66, 1>
        <<<4096, 256, 64 * 512 * 2, stream>>>(h1p, wp2, b2, h2p);
    // conv3: 128 oc, 131072 px, 64 px/block -> 2048 blocks, A via L2,
    // writes unpadded h3
    conv_s2_wmma<64, 128, 32, 32, 8, 1, false, 32, 32, 0>
        <<<2048, 256, 0, stream>>>(h2p, wp3, b3, h3);

    // Head
    meanpool_kernel<<<128, 256, 0, stream>>>(h3, h);
    latent_kernel<<<128, 256, 0, stream>>>(h, Wlat, blat, ze);
    vq_kernel<<<128, 128, 0, stream>>>(ze, cb, zq, lpart);
    loss_reduce_kernel<<<1, 128, 0, stream>>>(lpart, loss);
    proj_kernel<<<128, 128, 0, stream>>>(zq, Wproj, bproj, zd);

    // Decoder
    dec1_kernel<<<128, 256, 0, stream>>>(zd, d1, bd1, g1);
    dec2_kernel<<<128, 256, 0, stream>>>(g1, d2, bd2, g2);
    dec3_kernel<<<128, 64, 0, stream>>>(g2, d3, bd3, xr);
}
